// NaViT_Encoder_22299470201529
// MI455X (gfx1250) — compile-verified, matrix operand order checked by reading, not run on hardware
//
#include <hip/hip_runtime.h>
#include <hip/hip_bf16.h>
#include <math.h>

#define L_   2
#define B_   4
#define N_   2048
#define DIM_ 768
#define H_   12
#define DH_  64
#define MLP_ 3072
#define MROWS (B_ * N_) /* 8192 */

typedef __bf16 bf16t;
typedef __attribute__((ext_vector_type(16))) __bf16 v16bf;
typedef __attribute__((ext_vector_type(8)))  float  v8f;
typedef uint32_t u32x4 __attribute__((ext_vector_type(4)));
typedef int      i32x8 __attribute__((ext_vector_type(8)));
typedef int      i32x4 __attribute__((ext_vector_type(4)));

union BFrag { v16bf v; uint4 u[2]; };

// Async global->LDS copy of 16 bytes (ASYNCcnt-tracked, no VGPR round trip).
__device__ __forceinline__ void async_b128(const void* gaddr, void* ldsaddr) {
    asm volatile("global_load_async_to_lds_b128 %0, %1, off"
                 :: "v"((uint32_t)(uintptr_t)ldsaddr),
                    "v"((uint64_t)(uintptr_t)gaddr)
                 : "memory");
}
__device__ __forceinline__ void wait_async0() {
    asm volatile("s_wait_asynccnt 0x0" ::: "memory");
}

// TDM: DMA a rows x cols (bf16) 2-D tile into LDS with 16B padding every row
// (row = 32 elems = 64B = 16 DWORDs; pad 4 DWORDs => 40-elem LDS row stride).
__device__ __forceinline__ void tdm_load_tile_a(const bf16t* gtile, uint32_t lds_off,
                                                int rowsLeft, int K) {
    uint64_t ga = (uint64_t)(uintptr_t)gtile;
    u32x4 g0;
    g0[0] = 1u;                                                   // count=1 (user D#)
    g0[1] = lds_off;                                              // lds_addr
    g0[2] = (uint32_t)ga;                                         // global_addr[31:0]
    g0[3] = (uint32_t)((ga >> 32) & 0x01FFFFFFu) | (2u << 30);    // addr[56:32] | type=2
    i32x8 g1;
    g1[0] = (int)((1u << 16) | (1u << 20) | (3u << 22) | (3u << 25)); // 2B, pad_en, int=16dw, amt=4dw
    g1[1] = (int)(((uint32_t)K & 0xFFFFu) << 16);                 // tensor_dim0[15:0]
    g1[2] = (int)((((uint32_t)K >> 16) & 0xFFFFu) |
                  (((uint32_t)rowsLeft & 0xFFFFu) << 16));        // td0 hi | td1 lo
    g1[3] = (int)(((((uint32_t)rowsLeft) >> 16) & 0xFFFFu) | (32u << 16)); // td1 hi | tile_dim0=32
    g1[4] = (int)128u;                                            // tile_dim1=128, tile_dim2=0
    g1[5] = (int)(uint32_t)K;                                     // tensor_dim0_stride lo
    g1[6] = 0;                                                    // stride hi | dim1_stride lo
    g1[7] = 0;
    i32x4 z4 = {0, 0, 0, 0};
#if __clang_major__ >= 23
    i32x8 z8 = {0, 0, 0, 0, 0, 0, 0, 0};
    __builtin_amdgcn_tensor_load_to_lds(g0, g1, z4, z4, z8, 0);
#else
    __builtin_amdgcn_tensor_load_to_lds(g0, g1, z4, z4, 0);
#endif
}

// ---------------------------------------------------------------- utilities
__global__ void k_f32_to_bf16(const float* __restrict__ s, bf16t* __restrict__ d, size_t n) {
    size_t i = (size_t)blockIdx.x * 256 + threadIdx.x;
    if (i < n) d[i] = (bf16t)s[i];
}

__global__ void k_gelu_bf16(const float* __restrict__ x, bf16t* __restrict__ y, size_t n) {
    size_t i = (size_t)blockIdx.x * 256 + threadIdx.x;
    if (i < n) {
        float v = x[i];
        y[i] = (bf16t)(0.5f * v * (1.0f + erff(v * 0.70710678118f)));
    }
}

__global__ void k_mask(const int* __restrict__ lengths, float* __restrict__ m) {
    int idx = blockIdx.x * 256 + threadIdx.x;
    if (idx >= B_ * N_) return;
    int b = idx / N_;
    m[idx] = ((idx % N_) < lengths[b]) ? 1.0f : 0.0f;
}

// ---------------------------------------------------------------- layernorm
__global__ __launch_bounds__(256)
void k_layernorm(const float* __restrict__ x, const float* __restrict__ g,
                 bf16t* __restrict__ ob, float* __restrict__ of, int dim) {
    __shared__ float red[256];
    const int row = blockIdx.x;
    const int tid = threadIdx.x;
    const float* xr = x + (size_t)row * dim;

    float s = 0.f;
    for (int c = tid; c < dim; c += 256) s += xr[c];
    red[tid] = s; __syncthreads();
    for (int o = 128; o > 0; o >>= 1) { if (tid < o) red[tid] += red[tid + o]; __syncthreads(); }
    const float mean = red[0] / (float)dim;
    __syncthreads();

    float s2 = 0.f;
    for (int c = tid; c < dim; c += 256) { float d = xr[c] - mean; s2 += d * d; }
    red[tid] = s2; __syncthreads();
    for (int o = 128; o > 0; o >>= 1) { if (tid < o) red[tid] += red[tid + o]; __syncthreads(); }
    const float rstd = rsqrtf(red[0] / (float)dim + 1e-5f);

    for (int c = tid; c < dim; c += 256) {
        float v = (xr[c] - mean) * rstd * g[c];
        if (ob) ob[(size_t)row * dim + c] = (bf16t)v;
        if (of) of[(size_t)row * dim + c] = v;
    }
}

// ------------------------------------------------- QK RMSNorm + 2D RoPE
__global__ __launch_bounds__(256)
void k_rmsnorm_rope(const float* __restrict__ src, int rowStride,
                    const float* __restrict__ g,
                    const int* __restrict__ h_idx, const int* __restrict__ w_idx,
                    bf16t* __restrict__ dst) {
    int idx = blockIdx.x * 256 + threadIdx.x;
    if (idx >= B_ * H_ * N_) return;
    int n = idx % N_;
    int h = (idx / N_) % H_;
    int b = idx / (N_ * H_);

    const float* xr = src + (size_t)(b * N_ + n) * rowStride + h * DH_;
    float v[DH_];
    float ss = 0.f;
    #pragma unroll
    for (int d = 0; d < DH_; ++d) { v[d] = xr[d]; ss += v[d] * v[d]; }
    float sc = 8.0f / fmaxf(sqrtf(ss), 1e-12f);   // sqrt(DH)=8
    #pragma unroll
    for (int d = 0; d < DH_; ++d) v[d] *= sc * g[h * DH_ + d];

    float hi = (float)h_idx[b * N_ + n];
    float wi = (float)w_idx[b * N_ + n];
    const float lf = 0.575646273f;                // ln(10000)/16
    #pragma unroll
    for (int t = 0; t < 16; ++t) {
        float invf = __expf(-lf * (float)t);
        float th = hi * invf, tw = wi * invf;
        float sh, ch, sw, cw;
        __sincosf(th, &sh, &ch);
        __sincosf(tw, &sw, &cw);
        float a0 = v[t], a1 = v[t + 16];
        v[t]      = a0 * ch - a1 * sh;
        v[t + 16] = a1 * ch + a0 * sh;
        float b0 = v[32 + t], b1v = v[48 + t];
        v[32 + t] = b0 * cw - b1v * sw;
        v[48 + t] = b1v * cw + b0 * sw;
    }
    bf16t* o = dst + (((size_t)(b * H_ + h)) * N_ + n) * DH_;
    #pragma unroll
    for (int d = 0; d < DH_; ++d) o[d] = (bf16t)v[d];
}

__global__ void k_v_pack(const float* __restrict__ kv, bf16t* __restrict__ vb) {
    int idx = blockIdx.x * 256 + threadIdx.x;   // over B*H*N*DH
    if (idx >= B_ * H_ * N_ * DH_) return;
    int d  = idx & 63;
    int n  = (idx >> 6) & (N_ - 1);
    int bh = idx >> 17;               // 64*2048 = 2^17
    int h  = bh % H_;
    int b  = bh / H_;
    vb[idx] = (bf16t)kv[(size_t)(b * N_ + n) * (2 * DIM_) + DIM_ + h * DH_ + d];
}

// ---------------------------------------------------------------- WMMA GEMM
// C[M,Nn] = A_bf16[M,K] x W_bf16[K,Nn] (+bias) (+resid). Block tile 128x128.
// A tile staged by the Tensor Data Mover; W tile transposed by VALU.
__global__ __launch_bounds__(256)
void k_gemm_bf16(const bf16t* __restrict__ A, const bf16t* __restrict__ W,
                 const float* __restrict__ bias, const float* __restrict__ resid,
                 float* __restrict__ C, int M, int K, int Nn) {
    __shared__ alignas(16) bf16t As[128][40];   // [m][k]  (40 = 32 + 16B TDM pad)
    __shared__ alignas(16) bf16t Ws[128][40];   // [n][k]  (transposed)
    const int tid  = threadIdx.x;
    const int wave = tid >> 5;
    const int lane = tid & 31;
    const int bm = blockIdx.x * 128;
    const int bn = blockIdx.y * 128;
    const int wm = (wave >> 1) * 32;   // 0..96
    const int wn = (wave & 1) * 64;    // 0,64
    const int laneMod = lane & 15;
    const int hi8 = (lane >> 4) * 8;   // 0 or 8
    const uint32_t asBase = (uint32_t)(uintptr_t)&As[0][0];

    v8f acc[2][4] = {};

    for (int kt = 0; kt < K; kt += 32) {
        // Tensor Data Mover: 128x32 bf16 A tile -> padded LDS rows (wave 0 issues)
        if (tid < 32)
            tdm_load_tile_a(A + (size_t)bm * K + kt, asBase, M - bm, K);

        // prefetch next k-step tiles into GL2
        if (kt + 32 < K) {
            __builtin_prefetch(A + (size_t)(bm + (tid >> 1)) * K + kt + 32, 0, 1);
            __builtin_prefetch(W + (size_t)(kt + 32 + (tid & 31)) * Nn + bn + (tid >> 5) * 16, 0, 1);
        }

        #pragma unroll
        for (int i = 0; i < 16; ++i) {   // W tile 32x128 -> transposed
            int e = tid + 256 * i;
            int k = e >> 7;
            int n = e & 127;
            Ws[n][k] = W[(size_t)(kt + k) * Nn + bn + n];
        }
        if (tid < 32) __builtin_amdgcn_s_wait_tensorcnt(0);
        __syncthreads();

        BFrag af[2], bfq[4];
        #pragma unroll
        for (int mi = 0; mi < 2; ++mi) {
            const bf16t* p = &As[wm + mi * 16 + laneMod][hi8];
            af[mi].u[0] = *(const uint4*)p;
            af[mi].u[1] = *(const uint4*)(p + 16);
        }
        #pragma unroll
        for (int ni = 0; ni < 4; ++ni) {
            const bf16t* p = &Ws[wn + ni * 16 + laneMod][hi8];
            bfq[ni].u[0] = *(const uint4*)p;
            bfq[ni].u[1] = *(const uint4*)(p + 16);
        }
        #pragma unroll
        for (int mi = 0; mi < 2; ++mi)
            #pragma unroll
            for (int ni = 0; ni < 4; ++ni)
                acc[mi][ni] = __builtin_amdgcn_wmma_f32_16x16x32_bf16(
                    false, af[mi].v, false, bfq[ni].v, (short)0, acc[mi][ni], false, false);
        __syncthreads();
    }

    // epilogue: uniform branches hoisted out of the store loops
    float bcol[4] = {0.f, 0.f, 0.f, 0.f};
    if (bias) {
        #pragma unroll
        for (int ni = 0; ni < 4; ++ni)
            bcol[ni] = bias[bn + wn + ni * 16 + laneMod];
    }
    if (resid) {
        #pragma unroll
        for (int mi = 0; mi < 2; ++mi)
            #pragma unroll
            for (int ni = 0; ni < 4; ++ni)
                #pragma unroll
                for (int e = 0; e < 8; ++e) {
                    size_t o = (size_t)(bm + wm + mi * 16 + e + hi8) * Nn
                             + (bn + wn + ni * 16 + laneMod);
                    C[o] = acc[mi][ni][e] + bcol[ni] + resid[o];
                }
    } else {
        #pragma unroll
        for (int mi = 0; mi < 2; ++mi)
            #pragma unroll
            for (int ni = 0; ni < 4; ++ni)
                #pragma unroll
                for (int e = 0; e < 8; ++e) {
                    size_t o = (size_t)(bm + wm + mi * 16 + e + hi8) * Nn
                             + (bn + wn + ni * 16 + laneMod);
                    C[o] = acc[mi][ni][e] + bcol[ni];
                }
    }
}

// ---------------------------------------------------------------- flash attention
// Grid: (N/64, H, B), 128 threads (4 waves). Each wave owns a 16-row stripe.
// Q/K tiles staged with async global->LDS copies (ASYNCcnt).
__global__ __launch_bounds__(128)
void k_flash_attn(const bf16t* __restrict__ qb, const bf16t* __restrict__ kb,
                  const bf16t* __restrict__ vb, const int* __restrict__ lengths,
                  float* __restrict__ outf) {
    __shared__ alignas(16) bf16t Qs[64][72];
    __shared__ alignas(16) bf16t Ks[64][72];
    __shared__ alignas(16) bf16t Vt[64][72];   // [d][j]
    __shared__ alignas(16) bf16t Ps[64][72];
    __shared__ float Sc[64][68];
    __shared__ float mrow[64], lrow[64], arow[64];

    const int tid = threadIdx.x, wave = tid >> 5, lane = tid & 31;
    const int laneMod = lane & 15, hi8 = (lane >> 4) * 8;
    const int r0 = blockIdx.x * 64;
    const int h  = blockIdx.y;
    const int b  = blockIdx.z;
    const int len = lengths[b];
    const size_t headOff = ((size_t)(b * H_ + h)) * N_ * DH_;
    const bf16t* Q  = qb + headOff;
    const bf16t* Kp = kb + headOff;
    const bf16t* Vp = vb + headOff;

    {   // Q tile via async copies
        int r = tid >> 1, c0 = (tid & 1) * 32;
        const bf16t* gq = Q + (size_t)(r0 + r) * DH_ + c0;
        #pragma unroll
        for (int c = 0; c < 32; c += 8)
            async_b128(gq + c, &Qs[r][c0 + c]);
    }
    if (tid < 64) { mrow[tid] = -INFINITY; lrow[tid] = 0.f; arow[tid] = 0.f; }
    wait_async0();
    __syncthreads();

    const int wm = wave * 16;
    BFrag aq[2];
    #pragma unroll
    for (int s = 0; s < 2; ++s) {
        const bf16t* p = &Qs[wm + laneMod][s * 32 + hi8];
        aq[s].u[0] = *(const uint4*)p;
        aq[s].u[1] = *(const uint4*)(p + 16);
    }
    v8f o[4] = {};

    for (int jt = 0; jt < len; jt += 64) {
        __syncthreads();   // previous iteration fully done with Ks/Vt/Ps
        {   // K tile via async copies
            int r = tid >> 1, c0 = (tid & 1) * 32;
            const bf16t* gk = Kp + (size_t)(jt + r) * DH_ + c0;
            #pragma unroll
            for (int c = 0; c < 32; c += 8)
                async_b128(gk + c, &Ks[r][c0 + c]);
        }
        #pragma unroll
        for (int i = 0; i < 32; ++i) {   // V tile transposed (VALU path)
            int e = tid + 128 * i;
            int j = e >> 6, d = e & 63;
            Vt[d][j] = Vp[(size_t)(jt + j) * DH_ + d];
        }
        wait_async0();
        __syncthreads();

        // S = Q * K^T (16x64 stripe per wave)
        v8f sacc[4] = {};
        #pragma unroll
        for (int ni = 0; ni < 4; ++ni)
            #pragma unroll
            for (int s = 0; s < 2; ++s) {
                BFrag bk;
                const bf16t* p = &Ks[ni * 16 + laneMod][s * 32 + hi8];
                bk.u[0] = *(const uint4*)p;
                bk.u[1] = *(const uint4*)(p + 16);
                sacc[ni] = __builtin_amdgcn_wmma_f32_16x16x32_bf16(
                    false, aq[s].v, false, bk.v, (short)0, sacc[ni], false, false);
            }
        #pragma unroll
        for (int ni = 0; ni < 4; ++ni)
            #pragma unroll
            for (int e = 0; e < 8; ++e)
                Sc[wm + e + hi8][ni * 16 + laneMod] = sacc[ni][e];
        __syncthreads();

        // online softmax (one row per thread, masked columns)
        if (tid < 64) {
            const int row = tid;
            const int jlim = min(64, len - jt);
            float mOld = mrow[row];
            float mx = mOld;
            for (int j = 0; j < jlim; ++j) mx = fmaxf(mx, Sc[row][j]);
            float alpha = (mOld > -1e37f) ? __expf(mOld - mx) : 0.f;
            float ls = 0.f;
            for (int j = 0; j < 64; ++j) {
                float p = (j < jlim) ? __expf(Sc[row][j] - mx) : 0.f;
                Ps[row][j] = (bf16t)p;
                ls += p;
            }
            mrow[row] = mx;
            lrow[row] = lrow[row] * alpha + ls;
            arow[row] = alpha;
        }
        __syncthreads();

        // rescale O, then O += P * V
        float av[8];
        #pragma unroll
        for (int e = 0; e < 8; ++e) av[e] = arow[wm + e + hi8];
        #pragma unroll
        for (int ni = 0; ni < 4; ++ni)
            #pragma unroll
            for (int e = 0; e < 8; ++e)
                o[ni][e] *= av[e];

        BFrag ap[2];
        #pragma unroll
        for (int s = 0; s < 2; ++s) {
            const bf16t* p = &Ps[wm + laneMod][s * 32 + hi8];
            ap[s].u[0] = *(const uint4*)p;
            ap[s].u[1] = *(const uint4*)(p + 16);
        }
        #pragma unroll
        for (int ni = 0; ni < 4; ++ni)
            #pragma unroll
            for (int s = 0; s < 2; ++s) {
                BFrag bv;
                const bf16t* p = &Vt[ni * 16 + laneMod][s * 32 + hi8];
                bv.u[0] = *(const uint4*)p;
                bv.u[1] = *(const uint4*)(p + 16);
                o[ni] = __builtin_amdgcn_wmma_f32_16x16x32_bf16(
                    false, ap[s].v, false, bv.v, (short)0, o[ni], false, false);
            }
    }

    float inv[8];
    #pragma unroll
    for (int e = 0; e < 8; ++e) {
        float l = lrow[wm + e + hi8];
        inv[e] = (l > 0.f) ? (1.f / l) : 0.f;
    }
    #pragma unroll
    for (int ni = 0; ni < 4; ++ni)
        #pragma unroll
        for (int e = 0; e < 8; ++e) {
            int row = r0 + wm + e + hi8;
            int col = h * DH_ + ni * 16 + laneMod;
            outf[(size_t)(b * N_ + row) * (H_ * DH_) + col] = o[ni][e] * inv[e];
        }
}

// ---------------------------------------------------------------- host driver
extern "C" void kernel_launch(void* const* d_in, const int* in_sizes, int n_in,
                              void* d_out, int out_size, void* d_ws, size_t ws_size,
                              hipStream_t stream) {
    (void)in_sizes; (void)n_in; (void)out_size; (void)ws_size;

    const float* patches    = (const float*)d_in[0];
    const float* pe_ln1_g   = (const float*)d_in[1];
    const float* pe_W       = (const float*)d_in[2];
    const float* pe_b       = (const float*)d_in[3];
    const float* pe_ln2_g   = (const float*)d_in[4];
    const float* attn_ln_g  = (const float*)d_in[5];
    const float* qn_g       = (const float*)d_in[6];
    const float* kn_g       = (const float*)d_in[7];
    const float* Wq         = (const float*)d_in[8];
    const float* Wkv        = (const float*)d_in[9];
    const float* Wo         = (const float*)d_in[10];
    const float* ff_ln_g    = (const float*)d_in[11];
    const float* W1         = (const float*)d_in[12];
    const float* b1         = (const float*)d_in[13];
    const float* W2         = (const float*)d_in[14];
    const float* b2         = (const float*)d_in[15];
    const float* final_ln_g = (const float*)d_in[16];
    const int*   h_idx      = (const int*)d_in[17];
    const int*   w_idx      = (const int*)d_in[18];
    const int*   lengths    = (const int*)d_in[19];

    char* ws = (char*)d_ws;
    size_t off = 0;
    auto alloc = [&](size_t bytes) -> char* {
        char* p = ws + off;
        off = (off + bytes + 255) & ~(size_t)255;
        return p;
    };

    const size_t szPeW = (size_t)DIM_ * DIM_;
    const size_t szWq  = (size_t)DIM_ * DIM_;
    const size_t szWkv = (size_t)DIM_ * 2 * DIM_;
    const size_t szWo  = (size_t)DIM_ * DIM_;
    const size_t szW1  = (size_t)DIM_ * MLP_;
    const size_t szW2  = (size_t)MLP_ * DIM_;

    bf16t* wb_peW = (bf16t*)alloc(szPeW * 2);
    bf16t* wb_Wq  = (bf16t*)alloc(L_ * szWq  * 2);
    bf16t* wb_Wkv = (bf16t*)alloc(L_ * szWkv * 2);
    bf16t* wb_Wo  = (bf16t*)alloc(L_ * szWo  * 2);
    bf16t* wb_W1  = (bf16t*)alloc(L_ * szW1  * 2);
    bf16t* wb_W2  = (bf16t*)alloc(L_ * szW2  * 2);

    float* x_f   = (float*)alloc((size_t)MROWS * DIM_ * 4);       // residual stream
    bf16t* xn_b  = (bf16t*)alloc((size_t)MROWS * DIM_ * 2);       // LN output / attn-out bf16
    float* q_f   = (float*)alloc((size_t)MROWS * DIM_ * 4);       // q proj / pe tmp / attn-out
    float* kv_f  = (float*)alloc((size_t)MROWS * 2 * DIM_ * 4);   // kv proj; reused as gelu bf16
    bf16t* h_b   = (bf16t*)kv_f;                                  // 8192*3072 bf16 == same bytes
    bf16t* qbuf  = (bf16t*)alloc((size_t)MROWS * DIM_ * 2);
    bf16t* kbuf  = (bf16t*)alloc((size_t)MROWS * DIM_ * 2);
    bf16t* vbuf  = (bf16t*)alloc((size_t)MROWS * DIM_ * 2);
    float* h_pre = (float*)alloc((size_t)MROWS * MLP_ * 4);

    auto cvt = [&](const float* s, bf16t* d, size_t n) {
        k_f32_to_bf16<<<(unsigned)((n + 255) / 256), 256, 0, stream>>>(s, d, n);
    };
    auto gemm = [&](const bf16t* A, const bf16t* Wt, const float* bias,
                    const float* resid, float* C, int Nn, int K) {
        dim3 g(MROWS / 128, Nn / 128);
        k_gemm_bf16<<<g, 256, 0, stream>>>(A, Wt, bias, resid, C, MROWS, K, Nn);
    };

    // weight precision conversion
    cvt(pe_W, wb_peW, szPeW);
    cvt(Wq,  wb_Wq,  L_ * szWq);
    cvt(Wkv, wb_Wkv, L_ * szWkv);
    cvt(Wo,  wb_Wo,  L_ * szWo);
    cvt(W1,  wb_W1,  L_ * szW1);
    cvt(W2,  wb_W2,  L_ * szW2);

    // patch embedding: LN -> Linear(+b) -> LN
    k_layernorm<<<MROWS, 256, 0, stream>>>(patches, pe_ln1_g, xn_b, nullptr, DIM_);
    gemm(xn_b, wb_peW, pe_b, nullptr, q_f, DIM_, DIM_);
    k_layernorm<<<MROWS, 256, 0, stream>>>(q_f, pe_ln2_g, nullptr, x_f, DIM_);

    const int ropeBlocks = (B_ * H_ * N_ + 255) / 256;
    for (int l = 0; l < L_; ++l) {
        k_layernorm<<<MROWS, 256, 0, stream>>>(x_f, attn_ln_g + l * DIM_, xn_b, nullptr, DIM_);
        gemm(xn_b, wb_Wq  + l * szWq,  nullptr, nullptr, q_f,  DIM_,     DIM_);
        gemm(xn_b, wb_Wkv + l * szWkv, nullptr, nullptr, kv_f, 2 * DIM_, DIM_);

        k_rmsnorm_rope<<<ropeBlocks, 256, 0, stream>>>(q_f,  H_ * DH_,
            qn_g + l * H_ * DH_, h_idx, w_idx, qbuf);
        k_rmsnorm_rope<<<ropeBlocks, 256, 0, stream>>>(kv_f, 2 * H_ * DH_,
            kn_g + l * H_ * DH_, h_idx, w_idx, kbuf);
        k_v_pack<<<(B_ * H_ * N_ * DH_) / 256, 256, 0, stream>>>(kv_f, vbuf);

        k_flash_attn<<<dim3(N_ / 64, H_, B_), 128, 0, stream>>>(qbuf, kbuf, vbuf, lengths, q_f);

        cvt(q_f, xn_b, (size_t)MROWS * DIM_);
        gemm(xn_b, wb_Wo + l * szWo, nullptr, x_f, x_f, DIM_, DIM_);   // out@Wo + x

        k_layernorm<<<MROWS, 256, 0, stream>>>(x_f, ff_ln_g + l * DIM_, xn_b, nullptr, DIM_);
        gemm(xn_b, wb_W1 + l * szW1, b1 + l * MLP_, nullptr, h_pre, MLP_, DIM_);
        k_gelu_bf16<<<(unsigned)(((size_t)MROWS * MLP_ + 255) / 256), 256, 0, stream>>>(
            h_pre, h_b, (size_t)MROWS * MLP_);
        gemm(h_b, wb_W2 + l * szW2, b2 + l * DIM_, x_f, x_f, DIM_, MLP_);  // h@W2 + b2 + x
    }

    k_layernorm<<<MROWS, 256, 0, stream>>>(x_f, final_ln_g, nullptr, (float*)d_out, DIM_);
    k_mask<<<(B_ * N_) / 256, 256, 0, stream>>>(lengths, (float*)d_out + (size_t)MROWS * DIM_);
}
